// PINN_61864708932297
// MI455X (gfx1250) — matrix operand order repaired and checked by reading, not compiled, not run on hardware
//
#include <hip/hip_runtime.h>

typedef _Float16 hv16 __attribute__((ext_vector_type(16)));
typedef _Float16 hv8  __attribute__((ext_vector_type(8)));
typedef float    fv8  __attribute__((ext_vector_type(8)));

#define INV_PI_001 0.0031830988618379067f   // 0.01 / pi

// ---------------- LDS layout (bytes) ----------------
#define OFF_WW    0          // 4 * 64*64 f16  = 32768 (hidden weights, zero padded)
#define OFF_WOUT  32768      // 112*64  f16    = 14336
#define OFF_WIRK  47104      // 112*128 f16    = 28672
#define OFF_W1    75776      // 64 f32
#define OFF_B1    76032      // 64 f32
#define OFF_BHID  76288      // 4*64 f32
#define OFF_BOUT  77312      // 112 f32 (pad to 77824)
#define OFF_WAVE  77824
#define WAVE_BYTES 17408     // aScr 3*16*64 f16 (6144) + fBuf 16*128 f16 (4096) + uBuf 16*112 f32 (7168)
#define SMEM_BYTES (OFF_WAVE + 8 * WAVE_BYTES)   // 217088 bytes < 320KB WGP LDS

// Branchless tanh. Prefer the CDNA5 native V_TANH_F32 when the toolchain exposes
// it; otherwise exp2/rcp based, fully straight-line (no EXEC divergence between
// WMMAs, co-executes with the XDL matrix pipe).
static __device__ __forceinline__ float fast_tanh(float x) {
#if __has_builtin(__builtin_amdgcn_tanhf)
  return __builtin_amdgcn_tanhf(x);
#elif __has_builtin(__builtin_amdgcn_tanh_f32)
  return __builtin_amdgcn_tanh_f32(x);
#else
  float ax = __builtin_fabsf(x);
  float e  = __builtin_amdgcn_exp2f(ax * -2.8853900817779268f);  // exp(-2|x|)
  float t  = (1.f - e) * __builtin_amdgcn_rcpf(1.f + e);
  return __builtin_copysignf(t, x);
#endif
}

static __device__ __forceinline__ fv8 wmma_f16(hv16 a, hv16 b, fv8 c) {
  // D = A(16x32 f16) x B(32x16 f16) + C(16x16 f32)
  return __builtin_amdgcn_wmma_f32_16x16x32_f16(false, a, false, b, (short)0, c,
                                                false, false);
}

// A fragment: 16-bit A 16x32. Lane l (l<16): row m=l, K = kb..kb+7 and kb+16..kb+23
// with kb = 32*kt; lanes 16-31: same rows, kb += 8. Rows stored row-major f16 in LDS.
static __device__ __forceinline__ hv16 load_fragA(const _Float16* row0, int rowStride,
                                                  int lane, int kt) {
  int m  = lane & 15;
  int kb = kt * 32 + ((lane & 16) ? 8 : 0);
  const _Float16* p = row0 + m * rowStride + kb;
  hv8 lo = *(const hv8*)(p);        // K = kb .. kb+7
  hv8 hi = *(const hv8*)(p + 16);   // K = kb+16 .. kb+23
  return __builtin_shufflevector(lo, hi, 0, 1, 2, 3, 4, 5, 6, 7,
                                 8, 9, 10, 11, 12, 13, 14, 15);
}

// B fragment: B[k][n] = W[n][k] (W stored [n][rowStride] f16, K zero padded).
// Lanes 0-15 hold K=kt*32+0..7 and +16..23 of column n = nt*16 + lane; lanes 16-31 hold K +8.
static __device__ __forceinline__ hv16 load_fragB(const _Float16* w, int rowStride,
                                                  int lane, int nt, int kt) {
  int n  = nt * 16 + (lane & 15);
  int kb = kt * 32 + ((lane & 16) ? 8 : 0);
  const _Float16* p = w + n * rowStride + kb;
  hv8 lo = *(const hv8*)(p);
  hv8 hi = *(const hv8*)(p + 16);
  return __builtin_shufflevector(lo, hi, 0, 1, 2, 3, 4, 5, 6, 7,
                                 8, 9, 10, 11, 12, 13, 14, 15);
}

__global__ __launch_bounds__(256) void pinn_irk_kernel(
    const float* __restrict__ xg, const float* __restrict__ dtg,
    const float* __restrict__ irkg,
    const float* __restrict__ W0g, const float* __restrict__ W1g,
    const float* __restrict__ W2g, const float* __restrict__ W3g,
    const float* __restrict__ W4g,
    const float* __restrict__ b0g, const float* __restrict__ b1g,
    const float* __restrict__ b2g, const float* __restrict__ b3g,
    const float* __restrict__ b4g,
    const float* __restrict__ Woutg, const float* __restrict__ boutg,
    float* __restrict__ outg, int nPoints) {
  extern __shared__ char smem[];
  _Float16* wW   = (_Float16*)(smem + OFF_WW);    // [4][64][64]
  _Float16* wOut = (_Float16*)(smem + OFF_WOUT);  // [112][64]
  _Float16* wIrk = (_Float16*)(smem + OFF_WIRK);  // [112][128]  wIrk[q][k] = IRK[q][k]
  float* w1s  = (float*)(smem + OFF_W1);
  float* b1s  = (float*)(smem + OFF_B1);
  float* bHid = (float*)(smem + OFF_BHID);        // [4][64]
  float* bOut = (float*)(smem + OFF_BOUT);        // [112]

  const int tid = threadIdx.x;
  const float* Whg[4] = {W1g, W2g, W3g, W4g};
  const float* bhg[4] = {b1g, b2g, b3g, b4g};

  // ---- stage weights into LDS (f16, zero padded so padded K never poisons WMMA) ----
#pragma unroll
  for (int l = 0; l < 4; ++l) {
    for (int i = tid; i < 64 * 64; i += 256) {
      int n = i >> 6, k = i & 63;
      wW[l * 4096 + i] = (_Float16)((n < 50 && k < 50) ? Whg[l][n * 50 + k] : 0.f);
    }
  }
  for (int i = tid; i < 112 * 64; i += 256) {
    int n = i >> 6, k = i & 63;
    wOut[i] = (_Float16)((n < 101 && k < 50) ? Woutg[n * 50 + k] : 0.f);
  }
  for (int i = tid; i < 112 * 128; i += 256) {
    int q = i >> 7, k = i & 127;
    wIrk[i] = (_Float16)((q < 101 && k < 100) ? irkg[q * 100 + k] : 0.f);
  }
  if (tid < 64) {
    w1s[tid] = (tid < 50) ? W0g[tid] : 0.f;
    b1s[tid] = (tid < 50) ? b0g[tid] : 0.f;
#pragma unroll
    for (int l = 0; l < 4; ++l) bHid[l * 64 + tid] = (tid < 50) ? bhg[l][tid] : 0.f;
  }
  if (tid < 112) bOut[tid] = (tid < 101) ? boutg[tid] : 0.f;
  __syncthreads();

  const int wave = tid >> 5, lane = tid & 31;
  char* wbase = smem + OFF_WAVE + wave * WAVE_BYTES;
  _Float16* aScr = (_Float16*)wbase;            // 3 streams x [16][64] f16
  _Float16* fBuf = (_Float16*)(wbase + 6144);   // [16][128] f16
  float*    uBuf = (float*)(wbase + 10240);     // [16][112] f32

  const long ptBase = (long)blockIdx.x * 128 + (long)wave * 16;
  if (ptBase >= nPoints) return;
  const float dt = dtg[0];
  const int n0 = lane & 15;
  const int mBase = (lane & 16) ? 8 : 0;

  // ---- stage this wave's 16 x-values through LDS (uBuf reused as temp) ----
  if (lane < 16) uBuf[lane] = xg[ptBase + lane];

  // ---- layer 1 (1 -> 50): a = tanh(w*x+b), a1 = (1-t^2)w, a2 = -2t(1-t^2)w^2 ----
  for (int i = lane; i < 16 * 64; i += 32) {
    int m = i >> 6, n = i & 63;
    float w = w1s[n];                       // zero for padded n -> all streams 0
    float z = w * uBuf[m] + b1s[n];
    float t = fast_tanh(z);
    float d = 1.f - t * t;
    aScr[i]        = (_Float16)t;
    aScr[1024 + i] = (_Float16)(d * w);
    aScr[2048 + i] = (_Float16)(-2.f * t * d * w * w);
  }

  // ---- hidden layers 2..5 (50 -> 50), three Taylor streams per layer ----
  for (int l = 0; l < 4; ++l) {
    hv16 A0[2], A1[2], A2[2];               // preload so in-place LDS update is safe
#pragma unroll
    for (int kt = 0; kt < 2; ++kt) {
      A0[kt] = load_fragA(aScr,        64, lane, kt);
      A1[kt] = load_fragA(aScr + 1024, 64, lane, kt);
      A2[kt] = load_fragA(aScr + 2048, 64, lane, kt);
    }
    const _Float16* Wl = wW + l * 4096;
    const float* bl = bHid + l * 64;
    for (int nt = 0; nt < 4; ++nt) {
      fv8 c0 = {}, c1 = {}, c2 = {};
#pragma unroll
      for (int kt = 0; kt < 2; ++kt) {
        hv16 B = load_fragB(Wl, 64, lane, nt, kt);  // one B feeds three WMMAs
        c0 = wmma_f16(A0[kt], B, c0);
        c1 = wmma_f16(A1[kt], B, c1);
        c2 = wmma_f16(A2[kt], B, c2);
      }
      int n = nt * 16 + n0;
      float bn = bl[n];
#pragma unroll
      for (int r = 0; r < 8; ++r) {
        int m = mBase + r;
        float t  = fast_tanh(c0[r] + bn);
        float d  = 1.f - t * t;
        float z1 = c1[r];
        float a1v = d * z1;
        float a2v = d * c2[r] - 2.f * t * z1 * a1v;
        aScr[m * 64 + n]        = (_Float16)t;     // padded cols stay exactly 0
        aScr[1024 + m * 64 + n] = (_Float16)a1v;
        aScr[2048 + m * 64 + n] = (_Float16)a2v;
      }
    }
  }

  // ---- output layer (50 -> 101): u, u_x, u_xx; f = -u*u_x + (0.01/pi)*u_xx ----
  {
    hv16 A0[2], A1[2], A2[2];
#pragma unroll
    for (int kt = 0; kt < 2; ++kt) {
      A0[kt] = load_fragA(aScr,        64, lane, kt);
      A1[kt] = load_fragA(aScr + 1024, 64, lane, kt);
      A2[kt] = load_fragA(aScr + 2048, 64, lane, kt);
    }
    for (int i = lane; i < 16 * 128; i += 32) fBuf[i] = (_Float16)0.f;  // zero pad K
    for (int nt = 0; nt < 7; ++nt) {
      fv8 c0 = {}, c1 = {}, c2 = {};
#pragma unroll
      for (int kt = 0; kt < 2; ++kt) {
        hv16 B = load_fragB(wOut, 64, lane, nt, kt);
        c0 = wmma_f16(A0[kt], B, c0);
        c1 = wmma_f16(A1[kt], B, c1);
        c2 = wmma_f16(A2[kt], B, c2);
      }
      int n = nt * 16 + n0;
      float bn = bOut[n];
#pragma unroll
      for (int r = 0; r < 8; ++r) {
        int m = mBase + r;
        float u = c0[r] + bn;
        uBuf[m * 112 + n] = u;                          // keep u in f32 for final combine
        float fv = -u * c1[r] + INV_PI_001 * c2[r];
        if (n < 100) fBuf[m * 128 + n] = (_Float16)fv;  // f uses cols [0,100) only
      }
    }
  }

  // ---- out = u - dt * (f @ IRK^T) ----
  {
    hv16 F[4];
#pragma unroll
    for (int kt = 0; kt < 4; ++kt) F[kt] = load_fragA(fBuf, 128, lane, kt);
    for (int qt = 0; qt < 7; ++qt) {
      fv8 acc = {};
#pragma unroll
      for (int kt = 0; kt < 4; ++kt) {
        hv16 B = load_fragB(wIrk, 128, lane, qt, kt);
        acc = wmma_f16(F[kt], B, acc);
      }
      int q = qt * 16 + n0;
      if (q < 101) {
#pragma unroll
        for (int r = 0; r < 8; ++r) {
          int m = mBase + r;
          outg[(ptBase + m) * 101 + q] = uBuf[m * 112 + q] - dt * acc[r];
        }
      }
    }
  }
}

extern "C" void kernel_launch(void* const* d_in, const int* in_sizes, int n_in,
                              void* d_out, int out_size, void* d_ws, size_t ws_size,
                              hipStream_t stream) {
  (void)n_in; (void)d_ws; (void)ws_size; (void)out_size;
  const float* xg   = (const float*)d_in[0];
  const float* dtg  = (const float*)d_in[1];
  const float* irkg = (const float*)d_in[2];
  const float* W0g  = (const float*)d_in[3];
  const float* W1g  = (const float*)d_in[4];
  const float* W2g  = (const float*)d_in[5];
  const float* W3g  = (const float*)d_in[6];
  const float* W4g  = (const float*)d_in[7];
  const float* b0g  = (const float*)d_in[8];
  const float* b1g  = (const float*)d_in[9];
  const float* b2g  = (const float*)d_in[10];
  const float* b3g  = (const float*)d_in[11];
  const float* b4g  = (const float*)d_in[12];
  const float* Wog  = (const float*)d_in[13];
  const float* bog  = (const float*)d_in[14];

  int nPoints = in_sizes[0];
  int blocks = (nPoints + 127) / 128;

  // Allow >default dynamic LDS (212KB of the 320KB WGP LDS); ignore result.
  (void)hipFuncSetAttribute((const void*)pinn_irk_kernel,
                            hipFuncAttributeMaxDynamicSharedMemorySize,
                            (int)SMEM_BYTES);

  pinn_irk_kernel<<<blocks, 256, SMEM_BYTES, stream>>>(
      xg, dtg, irkg, W0g, W1g, W2g, W3g, W4g,
      b0g, b1g, b2g, b3g, b4g, Wog, bog,
      (float*)d_out, nPoints);
}
